// PureSNN_4045859193037
// MI455X (gfx1250) — compile-verified, matrix operand order checked by reading, not run on hardware
//
#include <hip/hip_runtime.h>

typedef __attribute__((ext_vector_type(16))) _Float16 v16h;
typedef __attribute__((ext_vector_type(8)))  float    v8f;

#define LEAK    0.8f
#define THRESH  0.5f
#define TSTEPS  16

union HPack { uint32_t u; _Float16 h[2]; };
union BMat  { v16h v; uint32_t u[8]; };

__global__ __launch_bounds__(256) void snn_wmma_kernel(
    const float* __restrict__ x,      // [B,2]
    const float* __restrict__ W_ih,   // [8,2]
    const float* __restrict__ b_ih,   // [8]
    const float* __restrict__ W_hh,   // [8,8]
    const float* __restrict__ b_hh,   // [8]
    const float* __restrict__ W_ho,   // [1,8]
    const float* __restrict__ b_ho,   // [1]
    float* __restrict__ out,          // [B,1]
    int B)
{
    const int tid  = blockIdx.x * blockDim.x + threadIdx.x;
    const int lane = threadIdx.x & 31;
    // B is 524288 (multiple of 256); clamp is a no-op safety net.
    const int s    = (tid < B) ? tid : (B - 1);

    // ---- per-lane constant input current: cur1 = W_ih @ x[s] + b_ih
    const float x0 = x[2 * s + 0];
    const float x1 = x[2 * s + 1];
    float cur1[8];
#pragma unroll
    for (int i = 0; i < 8; ++i)
        cur1[i] = fmaf(x0, W_ih[2 * i + 0], fmaf(x1, W_ih[2 * i + 1], b_ih[i]));

    // ---- A matrix (built once): block-diagonal W_hh, 16x32 f16.
    // lanes 0-7  : row M=lane,    K=0..7  -> W_hh[lane][k]
    // lanes 24-31: row M=lane-16, K=8..15 -> W_hh[lane-24][k]
    // all other lanes / all K>=16 halves: 0   (this also makes B's
    // elements 8-15 (K>=16) irrelevant: those multiply only zero A columns)
    const bool arow = (lane < 8) || (lane >= 24);
    const int  r    = lane & 7;
    v16h a;
#pragma unroll
    for (int k = 0; k < 8; ++k)
        a[k] = arow ? (_Float16)W_hh[8 * r + k] : (_Float16)0.0f;
#pragma unroll
    for (int k = 8; k < 16; ++k)
        a[k] = (_Float16)0.0f;

    // ---- per-lane broadcast weights
    float bhh[8], who[8];
#pragma unroll
    for (int i = 0; i < 8; ++i) { bhh[i] = b_hh[i]; who[i] = W_ho[i]; }
    const float bho = b_ho[0];

    // ---- state
    float m1[8];
#pragma unroll
    for (int i = 0; i < 8; ++i) m1[i] = 0.0f;
    v8f   m2 = {};
    float mo = 0.0f, tot = 0.0f;

    // Layer-1 step: leak + spike + reset, spikes packed as f16 pairs.
    // Independent of m2/WMMA output -> usable to fill the WMMA shadow.
    auto layer1_step = [&](uint32_t pk[4]) {
#pragma unroll
        for (int v = 0; v < 4; ++v) {
            HPack hp;
#pragma unroll
            for (int h = 0; h < 2; ++h) {
                const int j = 2 * v + h;
                m1[j] = fmaf(LEAK, m1[j], cur1[j]);
                const bool sp = (m1[j] >= THRESH);
                hp.h[h] = sp ? (_Float16)1.0f : (_Float16)0.0f;
                m1[j] = sp ? 0.0f : m1[j];
            }
            pk[v] = hp.u;
        }
    };

    uint32_t p[4];
    layer1_step(p);   // spikes for t = 0

#pragma unroll
    for (int t = 0; t < TSTEPS; ++t) {
        // ---- issue all 4 cross-half swizzles up front (one dscnt wait later)
        uint32_t q[4];
#pragma unroll
        for (int v = 0; v < 4; ++v)
            q[v] = (uint32_t)__builtin_amdgcn_ds_swizzle((int)p[v], 0x401F); // SWAPX16

        // ---- independent work in the DS shadow: C = leak*m2 + b_hh
        v8f c;
#pragma unroll
        for (int i = 0; i < 8; ++i)
            c[i] = fmaf(LEAK, m2[i], bhh[i]);

        // ---- assemble B (32x16 f16): elements 0-7 = own spikes (K0..7 in lanes 0-15,
        //      K8..15 in 16-31), elements 8-15 = partner lane's spikes. No masking needed: those
        //      feed only K>=16, whose A columns are identically zero, and all
        //      spike packs are finite (0.0h / 1.0h).
        BMat bm;
#pragma unroll
        for (int v = 0; v < 4; ++v) {
            bm.u[v]     = p[v];
            bm.u[4 + v] = q[v];
        }

        // ---- D = W_hh_blockdiag * s1 + (leak*m2 + b_hh)   (one WMMA)
        const v8f d = __builtin_amdgcn_wmma_f32_16x16x32_f16(
            /*neg_a=*/false, a, /*neg_b=*/false, bm.v,
            /*c_mod=*/(short)0, c, /*reuse_a=*/false, /*reuse_b=*/false);

        // ---- software pipeline: next step's layer 1 is independent of d,
        //      fills the WMMA->VALU hazard window.
        if (t + 1 < TSTEPS)
            layer1_step(p);

        // ---- layer 2 spike/reset + readout dot (in-lane, 8 wide)
        float acc = bho;
        v8f m2n;
#pragma unroll
        for (int i = 0; i < 8; ++i) {
            const bool sp = (d[i] >= THRESH);
            acc   += sp ? who[i] : 0.0f;
            m2n[i] = sp ? 0.0f : d[i];
        }
        m2 = m2n;

        mo = fmaf(LEAK, mo, acc);
        const bool spo = (mo >= THRESH);
        tot += spo ? 1.0f : 0.0f;
        mo = spo ? 0.0f : mo;
    }

    if (tid < B)
        out[tid] = tot * (1.0f / (float)TSTEPS);
}

extern "C" void kernel_launch(void* const* d_in, const int* in_sizes, int n_in,
                              void* d_out, int out_size, void* d_ws, size_t ws_size,
                              hipStream_t stream) {
    const float* x    = (const float*)d_in[0];
    const float* W_ih = (const float*)d_in[1];
    const float* b_ih = (const float*)d_in[2];
    const float* W_hh = (const float*)d_in[3];
    const float* b_hh = (const float*)d_in[4];
    const float* W_ho = (const float*)d_in[5];
    const float* b_ho = (const float*)d_in[6];
    float* out = (float*)d_out;

    const int B     = in_sizes[0] / 2;   // x is [B,2]
    const int block = 256;               // 8 wave32s, 32 samples per wave
    const int grid  = (B + block - 1) / block;

    snn_wmma_kernel<<<grid, block, 0, stream>>>(x, W_ih, b_ih, W_hh, b_hh,
                                                W_ho, b_ho, out, B);
}
